// SelfAttention_91311004713583
// MI455X (gfx1250) — matrix operand order repaired and checked by reading, not compiled, not run on hardware
//
#include <hip/hip_runtime.h>

// ---------------------------------------------------------------------------
// MI455X (gfx1250) self-attention block, wave32 + v_wmma_f32_16x16x32_f16.
// Compute-bound (~137 GFLOP vs ~60MB weights) -> all GEMMs on WMMA, flash-
// attention middle kernel so probs are never materialized. K-chunk staging
// uses global_load_async_to_lds_b128 (ASYNCcnt) — the CDNA5 async path.
// ---------------------------------------------------------------------------

typedef __attribute__((ext_vector_type(16))) _Float16 v16h;
typedef __attribute__((ext_vector_type(8)))  _Float16 v8h;
typedef __attribute__((ext_vector_type(8)))  float    v8f;

#define B_SZ   4
#define L_SEQ  2048
#define C_DIM  1024
#define HEADS  16
#define DHEAD  64
#define MAX_SCALE_MUL 4.60517018598809136804f

static __device__ __forceinline__ v16h cat8(v8h lo, v8h hi) {
  return __builtin_shufflevector(lo, hi, 0,1,2,3,4,5,6,7,8,9,10,11,12,13,14,15);
}
static __device__ __forceinline__ v8f wmma32f16(v16h a, v16h b, v8f c) {
  // (neg_a, A, neg_b, B, c_mod, C, reuse_a, reuse_b)
  return __builtin_amdgcn_wmma_f32_16x16x32_f16(false, a, false, b, (short)0, c,
                                                false, false);
}

// ---------------------------------------------------------------------------
// Kernel 1: QKV = x @ qkv_w^T (+q/v bias), scatter f16 into [B,H,L,D] buffers.
// Block tile 128x128, K-step 32. 8 waves in a 2x4 grid, each wave 64x32.
// N-tile (128) never spans a part boundary (1024 % 128 == 0), so part/dst/bias
// are block-uniform -> branchless epilogue.
// ---------------------------------------------------------------------------
__global__ __launch_bounds__(256) void qkv_gemm_kernel(
    const float* __restrict__ x, const float* __restrict__ w,
    const float* __restrict__ q_bias, const float* __restrict__ v_bias,
    _Float16* __restrict__ qbuf, _Float16* __restrict__ kbuf,
    _Float16* __restrict__ vbuf)
{
  constexpr int BM = 128, BN = 128, BK = 32;
  __shared__ _Float16 as_[BM * BK];
  __shared__ _Float16 bs_[BN * BK];

  const int tiles_n = (3 * C_DIM) / BN;          // 24
  const int tm = blockIdx.x / tiles_n;
  const int tn = blockIdx.x % tiles_n;
  const int m0 = tm * BM, n0 = tn * BN;
  const int tid  = threadIdx.x;
  const int lane = tid & 31, wid = tid >> 5;
  const int wm = (wid >> 2) * 64, wn = (wid & 3) * 32;
  const int lm = lane & 15, lk = lane >> 4;
  const int lr = tid >> 1;                       // tile row this thread stages
  const int lc = (tid & 1) << 4;                 // 0 or 16

  v8f acc[4][2] = {};

  for (int k0 = 0; k0 < C_DIM; k0 += BK) {
    __syncthreads();
    {
      const float4* ga = (const float4*)(x + (size_t)(m0 + lr) * C_DIM + k0 + lc);
      const float4* gb = (const float4*)(w + (size_t)(n0 + lr) * C_DIM + k0 + lc);
      float fa[16], fb[16];
      *(float4*)(fa + 0)  = ga[0]; *(float4*)(fa + 4)  = ga[1];
      *(float4*)(fa + 8)  = ga[2]; *(float4*)(fa + 12) = ga[3];
      *(float4*)(fb + 0)  = gb[0]; *(float4*)(fb + 4)  = gb[1];
      *(float4*)(fb + 8)  = gb[2]; *(float4*)(fb + 12) = gb[3];
      _Float16* pa = as_ + lr * BK + lc;
      _Float16* pb = bs_ + lr * BK + lc;
#pragma unroll
      for (int i = 0; i < 16; i++) { pa[i] = (_Float16)fa[i]; pb[i] = (_Float16)fb[i]; }
    }
    __syncthreads();

    v16h afr[4];
#pragma unroll
    for (int mi = 0; mi < 4; mi++) {
      const v8h* pr = (const v8h*)(as_ + (wm + mi * 16 + lm) * BK);
      afr[mi] = cat8(pr[2 * lk], pr[2 * lk + 1]); // K = lk*16..
    }
    v16h bfr[2];
#pragma unroll
    for (int ni = 0; ni < 2; ni++)
      bfr[ni] = ((const v16h*)(bs_ + (wn + ni * 16 + lm) * BK))[lk];  // K = lk*16..

#pragma unroll
    for (int mi = 0; mi < 4; mi++)
#pragma unroll
      for (int ni = 0; ni < 2; ni++)
        acc[mi][ni] = wmma32f16(afr[mi], bfr[ni], acc[mi][ni]);
  }

  // Branchless epilogue: part is block-uniform; bias depends only on (ni, lane).
  const int part = n0 >> 10;                         // 0=q, 1=k, 2=v
  _Float16* const dst = (part == 0) ? qbuf : (part == 1) ? kbuf : vbuf;
  const int hdb = (n0 & (C_DIM - 1)) + wn + lm;      // head*64+d for ni=0
  float bias[2] = {0.f, 0.f};
  if (part == 0) { bias[0] = q_bias[hdb]; bias[1] = q_bias[hdb + 16]; }
  else if (part == 2) { bias[0] = v_bias[hdb]; bias[1] = v_bias[hdb + 16]; }

#pragma unroll
  for (int ni = 0; ni < 2; ni++) {
    const int hd = hdb + ni * 16;
    const int h = hd >> 6, d = hd & (DHEAD - 1);
#pragma unroll
    for (int mi = 0; mi < 4; mi++) {
#pragma unroll
      for (int r = 0; r < 8; r++) {
        int gm = m0 + wm + mi * 16 + r + 8 * lk;     // C/D layout: row = r+8*hi
        int bb = gm >> 11, l = gm & (L_SEQ - 1);
        dst[(((size_t)(bb * HEADS + h)) * L_SEQ + l) * DHEAD + d] =
            (_Float16)(acc[mi][ni][r] + bias[ni]);
      }
    }
  }
}

// ---------------------------------------------------------------------------
// Kernel 2: in-place L2-norm of q (with exp(min(scale_mul,ln100)) scale) and k.
// One thread per (b,h,l) row of D=64.
// ---------------------------------------------------------------------------
__global__ __launch_bounds__(256) void norm_kernel(
    _Float16* __restrict__ qbuf, _Float16* __restrict__ kbuf,
    const float* __restrict__ scale_mul)
{
  const int nrows = B_SZ * HEADS * L_SEQ;        // 131072
  int r = blockIdx.x * blockDim.x + threadIdx.x; // 0 .. 2*nrows
  bool isq = r < nrows;
  int row = isq ? r : r - nrows;
  _Float16* p = (isq ? qbuf : kbuf) + (size_t)row * DHEAD;

  v8h v[8];
#pragma unroll
  for (int j = 0; j < 8; j++) v[j] = ((const v8h*)p)[j];
  float s = 0.f;
#pragma unroll
  for (int j = 0; j < 8; j++)
#pragma unroll
    for (int i = 0; i < 8; i++) { float f = (float)v[j][i]; s += f * f; }
  float sc = 1.0f / fmaxf(sqrtf(s), 1e-12f);
  if (isq) {
    int h = (row >> 11) & (HEADS - 1);
    sc *= __expf(fminf(scale_mul[h], MAX_SCALE_MUL));
  }
#pragma unroll
  for (int j = 0; j < 8; j++) {
    v8h o;
#pragma unroll
    for (int i = 0; i < 8; i++) o[i] = (_Float16)((float)v[j][i] * sc);
    ((v8h*)p)[j] = o;
  }
}

// ---------------------------------------------------------------------------
// Kernel 3: flash attention. Block = one (b,h) x 128 q-rows; wave = 16 q-rows.
// K-chunk staged via global_load_async_to_lds_b128 (ASYNCcnt); V staged
// transposed through VGPRs. Per 32-key chunk: S = Q@K^T (4 wmma), online
// softmax (shfl_xor over 16-lane N-group), P relaid via wave-private LDS,
// O += P@V (4 wmma).
// ---------------------------------------------------------------------------
__global__ __launch_bounds__(256) void attn_kernel(
    const _Float16* __restrict__ qbuf, const _Float16* __restrict__ kbuf,
    const _Float16* __restrict__ vbuf, const float* __restrict__ attn_bias,
    _Float16* __restrict__ abuf)
{
  __shared__ _Float16 ks_[32 * 64];      // [key][d]
  __shared__ _Float16 vsT[64 * 32];      // [d][key] (transposed stage)
  __shared__ _Float16 ps[8][16 * 32];    // per-wave P scratch [m][key]

  const int bh = blockIdx.x >> 4;        // 0..63
  const int qt = blockIdx.x & 15;
  const int b = bh >> 4, h = bh & (HEADS - 1);
  const int tid = threadIdx.x, lane = tid & 31, wid = tid >> 5;
  const int lm = lane & 15, lk = lane >> 4;

  const int q0 = qt * 128 + wid * 16;    // q-row base (within L) for this wave
  const _Float16* qrow = qbuf + ((size_t)bh * L_SEQ + q0) * DHEAD;

  // Q fragments resident in registers: A-matrix 16x32 per 32-wide d-step.
  v16h qfr[2];
#pragma unroll
  for (int ks = 0; ks < 2; ks++) {
    const v8h* pq = (const v8h*)(qrow + lm * DHEAD + ks * 32);
    qfr[ks] = cat8(pq[2 * lk], pq[2 * lk + 1]);
  }

  v8f o[4] = {};
  float mrow[8], lrow[8];
#pragma unroll
  for (int r = 0; r < 8; r++) { mrow[r] = -1e30f; lrow[r] = 0.f; }

  const _Float16* kbase = kbuf + (size_t)bh * L_SEQ * DHEAD;
  const _Float16* vbase = vbuf + (size_t)bh * L_SEQ * DHEAD;

  const int sr = tid >> 3;               // staging: key row 0..31
  const int sc = (tid & 7) << 3;         // staging: d col 0,8,..,56
  const unsigned kdst = (unsigned)(uintptr_t)(ks_ + sr * 64 + sc);
  const float* brow = attn_bias + (size_t)(q0 + 8 * lk) * L_SEQ + lm;

  for (int kc = 0; kc < L_SEQ; kc += 32) {
    __syncthreads();
    // K chunk: 16B/lane async copy straight into LDS (no VGPR round-trip).
    {
      const _Float16* gk = kbase + (size_t)(kc + sr) * DHEAD + sc;
      asm volatile("global_load_async_to_lds_b128 %0, %1, off"
                   :: "v"(kdst), "v"(gk) : "memory");
    }
    // V chunk: stage transposed so P@V B-fragments are contiguous v16h reads.
    v8h vv = *(const v8h*)(vbase + (size_t)(kc + sr) * DHEAD + sc);
#pragma unroll
    for (int j = 0; j < 8; j++) vsT[(sc + j) * 32 + sr] = vv[j];
    asm volatile("s_wait_asynccnt 0" ::: "memory");
    __syncthreads();

    // S = Q @ K^T for 2 key sub-tiles of 16.
    v8f s[2] = {};
#pragma unroll
    for (int ni = 0; ni < 2; ni++) {
#pragma unroll
      for (int ks2 = 0; ks2 < 2; ks2++) {
        v16h bf = ((const v16h*)(ks_ + (ni * 16 + lm) * 64 + ks2 * 32))[lk];
        s[ni] = wmma32f16(qfr[ks2], bf, s[ni]);
      }
#pragma unroll
      for (int r = 0; r < 8; r++)
        s[ni][r] += brow[(size_t)r * L_SEQ + kc + ni * 16];
    }

    // Online softmax: rows live across the 16-lane group, one row per vgpr.
#pragma unroll
    for (int r = 0; r < 8; r++) {
      float t = fmaxf(s[0][r], s[1][r]);
      t = fmaxf(t, __shfl_xor(t, 1));
      t = fmaxf(t, __shfl_xor(t, 2));
      t = fmaxf(t, __shfl_xor(t, 4));
      t = fmaxf(t, __shfl_xor(t, 8));
      float nm = fmaxf(mrow[r], t);
      float alpha = __expf(mrow[r] - nm);
      mrow[r] = nm;
      float e0 = __expf(s[0][r] - nm);
      float e1 = __expf(s[1][r] - nm);
      float rs = e0 + e1;
      rs += __shfl_xor(rs, 1);
      rs += __shfl_xor(rs, 2);
      rs += __shfl_xor(rs, 4);
      rs += __shfl_xor(rs, 8);
      lrow[r] = lrow[r] * alpha + rs;
      ps[wid][(r + 8 * lk) * 32 + lm]      = (_Float16)e0;
      ps[wid][(r + 8 * lk) * 32 + 16 + lm] = (_Float16)e1;
#pragma unroll
      for (int di = 0; di < 4; di++) o[di][r] *= alpha;
    }

    // Re-layout P through wave-private LDS (LDS ops are in-order per wave).
    asm volatile("s_wait_dscnt 0" ::: "memory");
    const v8h* pp = (const v8h*)(&ps[wid][lm * 32]);
    v16h pf = cat8(pp[2 * lk], pp[2 * lk + 1]);

    // O += P @ V over 4 d sub-tiles.
#pragma unroll
    for (int di = 0; di < 4; di++) {
      v16h vf = ((const v16h*)(vsT + (di * 16 + lm) * 32))[lk];
      o[di] = wmma32f16(pf, vf, o[di]);
    }
  }

  // Final 1/l scale, write f16 [B,L,C] with C index = h*64 + d.
#pragma unroll
  for (int r = 0; r < 8; r++) {
    float inv = 1.0f / lrow[r];
    int l = q0 + r + 8 * lk;
#pragma unroll
    for (int di = 0; di < 4; di++) {
      int d = di * 16 + lm;
      abuf[((size_t)(b * L_SEQ + l)) * C_DIM + h * DHEAD + d] =
          (_Float16)(o[di][r] * inv);
    }
  }
}

// ---------------------------------------------------------------------------
// Kernel 4: out = attn @ proj_w^T + proj_b (f16 A, f32->f16 B, f32 out).
// proj_b depends only on (ni, lane) -> hoisted out of the store loops.
// ---------------------------------------------------------------------------
__global__ __launch_bounds__(256) void proj_gemm_kernel(
    const _Float16* __restrict__ a, const float* __restrict__ w,
    const float* __restrict__ proj_b, float* __restrict__ out)
{
  constexpr int BM = 128, BN = 128, BK = 32;
  __shared__ _Float16 as_[BM * BK];
  __shared__ _Float16 bs_[BN * BK];

  const int tm = blockIdx.x >> 3;        // C_DIM/BN == 8 n-tiles
  const int tn = blockIdx.x & 7;
  const int m0 = tm * BM, n0 = tn * BN;
  const int tid  = threadIdx.x;
  const int lane = tid & 31, wid = tid >> 5;
  const int wm = (wid >> 2) * 64, wn = (wid & 3) * 32;
  const int lm = lane & 15, lk = lane >> 4;
  const int lr = tid >> 1;
  const int lc = (tid & 1) << 4;

  v8f acc[4][2] = {};

  for (int k0 = 0; k0 < C_DIM; k0 += BK) {
    __syncthreads();
    {
      const v8h* ga = (const v8h*)(a + (size_t)(m0 + lr) * C_DIM + k0 + lc);
      v8h* pa = (v8h*)(as_ + lr * BK + lc);
      pa[0] = ga[0]; pa[1] = ga[1];
      const float4* gb = (const float4*)(w + (size_t)(n0 + lr) * C_DIM + k0 + lc);
      float fb[16];
      *(float4*)(fb + 0)  = gb[0]; *(float4*)(fb + 4)  = gb[1];
      *(float4*)(fb + 8)  = gb[2]; *(float4*)(fb + 12) = gb[3];
      _Float16* pb = bs_ + lr * BK + lc;
#pragma unroll
      for (int i = 0; i < 16; i++) pb[i] = (_Float16)fb[i];
    }
    __syncthreads();

    v16h afr[4];
#pragma unroll
    for (int mi = 0; mi < 4; mi++) {
      const v8h* pr = (const v8h*)(as_ + (wm + mi * 16 + lm) * BK);
      afr[mi] = cat8(pr[2 * lk], pr[2 * lk + 1]);
    }
    v16h bfr[2];
#pragma unroll
    for (int ni = 0; ni < 2; ni++)
      bfr[ni] = ((const v16h*)(bs_ + (wn + ni * 16 + lm) * BK))[lk];

#pragma unroll
    for (int mi = 0; mi < 4; mi++)
#pragma unroll
      for (int ni = 0; ni < 2; ni++)
        acc[mi][ni] = wmma32f16(afr[mi], bfr[ni], acc[mi][ni]);
  }

  const int gn0 = n0 + wn + lm;
  const float pb0 = proj_b[gn0], pb1 = proj_b[gn0 + 16];
#pragma unroll
  for (int mi = 0; mi < 4; mi++) {
#pragma unroll
    for (int r = 0; r < 8; r++) {
      int gm = m0 + wm + mi * 16 + r + 8 * lk;
      out[(size_t)gm * C_DIM + gn0]      = acc[mi][0][r] + pb0;
      out[(size_t)gm * C_DIM + gn0 + 16] = acc[mi][1][r] + pb1;
    }
  }
}

// ---------------------------------------------------------------------------
extern "C" void kernel_launch(void* const* d_in, const int* in_sizes, int n_in,
                              void* d_out, int out_size, void* d_ws, size_t ws_size,
                              hipStream_t stream) {
  const float* x         = (const float*)d_in[0];
  const float* attn_bias = (const float*)d_in[1];
  const float* qkv_w     = (const float*)d_in[2];
  const float* q_bias    = (const float*)d_in[3];
  const float* v_bias    = (const float*)d_in[4];
  const float* scale_mul = (const float*)d_in[5];
  const float* proj_w    = (const float*)d_in[6];
  const float* proj_b    = (const float*)d_in[7];
  float* out = (float*)d_out;

  const size_t nelem = (size_t)B_SZ * HEADS * L_SEQ * DHEAD;  // 8388608
  _Float16* ws   = (_Float16*)d_ws;
  _Float16* qbuf = ws;
  _Float16* kbuf = qbuf + nelem;
  _Float16* vbuf = kbuf + nelem;
  _Float16* abuf = vbuf + nelem;   // [B,L,C] f16

  // 1) QKV projection: M=8192, N=3072, K=1024 -> 64*24 tiles.
  qkv_gemm_kernel<<<64 * 24, 256, 0, stream>>>(x, qkv_w, q_bias, v_bias,
                                               qbuf, kbuf, vbuf);
  // 2) L2-normalize q (+scale) and k: 262144 rows total.
  norm_kernel<<<(2 * B_SZ * HEADS * L_SEQ) / 256, 256, 0, stream>>>(
      qbuf, kbuf, scale_mul);
  // 3) Flash attention: B*H * (L/128) = 1024 blocks.
  attn_kernel<<<B_SZ * HEADS * (L_SEQ / 128), 256, 0, stream>>>(
      qbuf, kbuf, vbuf, attn_bias, abuf);
  // 4) Output projection: M=8192, N=1024, K=1024 -> 64*8 tiles.
  proj_gemm_kernel<<<64 * 8, 256, 0, stream>>>(abuf, proj_w, proj_b, out);
}
